// FermiNet_22823456211613
// MI455X (gfx1250) — matrix operand order repaired and checked by reading, not compiled, not run on hardware
//
#include <hip/hip_runtime.h>
#include <hip/hip_bf16.h>

// ---------------------------------------------------------------------------
// FermiNet forward for gfx1250 (MI455X): bf16 WMMA GEMMs + wave-parallel LU.
// ---------------------------------------------------------------------------

typedef __attribute__((ext_vector_type(16))) __bf16 v16bf;
typedef __attribute__((ext_vector_type(8)))  float  v8f;

// ---------------- weight prep: fp32 (K x N) -> bf16 transposed (N x Kpad) ---
__global__ void k_prep_bf16t(const float* __restrict__ W, int K, int N, int Kpad,
                             __bf16* __restrict__ out) {
    int total = N * Kpad;
    for (int idx = blockIdx.x * blockDim.x + threadIdx.x; idx < total;
         idx += gridDim.x * blockDim.x) {
        int n = idx / Kpad;
        int k = idx - n * Kpad;
        float v = (k < K) ? W[(size_t)k * N + n] : 0.0f;
        out[idx] = (__bf16)v;
    }
}

// ---------------- pointers bundle ------------------------------------------
struct FNParams {
    const float*  r;        // (B, 32, 3)
    const float*  a;        // (3,)
    const __bf16* Wst[4];   // s-layer weights, transposed bf16 (256 x K)
    const float*  bs[4];
    const __bf16* Wpt[4];   // p-layer weights, transposed bf16 (32 x 32)
    const float*  bp[4];
    const __bf16* Wwt;      // (256 x 832)
    const float*  bw;
    const __bf16* Wwut;     // (256 x 256)
    const float*  bwu;
    const __bf16* Wwdt;     // (256 x 256)
    const float*  bwd;
    const float*  euw;      // (3 x 256) row-major
    const float*  eub;
    const float*  edw;
    const float*  edb;
    float*        out;
};

// ---------------- generic bf16 WMMA GEMM over tiles ------------------------
// A: M x K (bf16, LDS), lda multiple of 32, 16B aligned rows
// Bt: N x K (bf16, global, transposed), ldb multiple of 32
// out: bf16 with tanh (out_f32==0) or raw f32 (out_f32==1)
__device__ inline void gemm_bf16(const __bf16* __restrict__ A, int lda,
                                 const __bf16* __restrict__ Bt, int ldb,
                                 const float* __restrict__ bias,
                                 int mtiles, int ntiles, int kblocks,
                                 void* outp, int ldo, int out_f32) {
    const int tid  = threadIdx.x;
    const int wave = tid >> 5;
    const int lane = tid & 31;
    const int half = lane >> 4;   // 0: lanes 0-15, 1: lanes 16-31
    const int lr   = lane & 15;
    const int tot  = mtiles * ntiles;
    for (int t = wave; t < tot; t += 8) {
        const int mt = t / ntiles;
        const int nt = t - mt * ntiles;
        v8f acc = {0.f, 0.f, 0.f, 0.f, 0.f, 0.f, 0.f, 0.f};
        // A fragment per ISA: lanes 0-15 K {0..7,16..23}, lanes 16-31 K {8..15,24..31}
        const __bf16* ap = A  + (size_t)(mt * 16 + lr) * lda + half * 8;
        // B fragment: same K order as A: lanes 0-15 K {0..7,16..23}, lanes 16-31 K {8..15,24..31} (col = lr)
        const __bf16* bp = Bt + (size_t)(nt * 16 + lr) * ldb + half * 8;
        for (int kb = 0; kb < kblocks; ++kb) {
            union { v16bf v; uint4 q[2]; } fa, fb;
            fa.q[0] = *(const uint4*)(ap);
            fa.q[1] = *(const uint4*)(ap + 16);
            fb.q[0] = *(const uint4*)(bp);
            fb.q[1] = *(const uint4*)(bp + 16);
            ap += 32;
            bp += 32;
            acc = __builtin_amdgcn_wmma_f32_16x16x32_bf16(
                false, fa.v, false, fb.v, (short)0, acc, false, false);
        }
        const int   n  = nt * 16 + lr;       // each lane owns one column
        const float bv = bias[n];
        if (out_f32) {
            float* o = (float*)outp;
#pragma unroll
            for (int rr = 0; rr < 8; ++rr) {
                int m = mt * 16 + half * 8 + rr;
                o[(size_t)m * ldo + n] = acc[rr] + bv;
            }
        } else {
            __bf16* o = (__bf16*)outp;
#pragma unroll
            for (int rr = 0; rr < 8; ++rr) {
                int m = mt * 16 + half * 8 + rr;
                o[(size_t)m * ldo + n] = (__bf16)tanhf(acc[rr] + bv);
            }
        }
    }
}

// ---------------- slogdet of a 16x16 matrix, 16 lanes (row-per-lane) -------
__device__ inline void slogdet16(const float* __restrict__ M, int l,
                                 float& sgn_out, float& log_out) {
    float v[16];
#pragma unroll
    for (int c = 0; c < 16; ++c) v[c] = M[l * 16 + c];
    float sign = 1.0f, ld = 0.0f;
#pragma unroll
    for (int k = 0; k < 16; ++k) {
        // argmax |v[k]| over rows >= k (butterfly over 16-lane group)
        float key  = (l >= k) ? fabsf(v[k]) : -1.0f;
        int   pidx = l;
#pragma unroll
        for (int off = 8; off > 0; off >>= 1) {
            float okey = __shfl_xor(key, off, 16);
            int   oi   = __shfl_xor(pidx, off, 16);
            if (okey > key) { key = okey; pidx = oi; }
        }
        const int p = pidx;
        if (p != k) sign = -sign;
        // swap rows k <-> p (shuffles executed by all lanes; writes predicated)
#pragma unroll
        for (int c = 0; c < 16; ++c) {
            float vk = __shfl(v[c], k, 16);
            float vp = __shfl(v[c], p, 16);
            if (p != k) {
                if (l == k) v[c] = vp;
                else if (l == p) v[c] = vk;
            }
        }
        float pk = __shfl(v[k], k, 16);
        ld += logf(fabsf(pk));
        if (pk < 0.0f) sign = -sign;
        float f = (l > k) ? v[k] / pk : 0.0f;
#pragma unroll
        for (int c = 0; c < 16; ++c) {
            float pc = __shfl(v[c], k, 16);
            v[c] -= f * pc;
        }
    }
    sgn_out = sign;
    log_out = ld;
}

// ---------------- LDS layout (dynamic) -------------------------------------
// 0       : sA   32 x 832 bf16  (concat features)          53248 B
// 53248   : sTmp 32 x 256 bf16  (new h_s; later w buffer)   16384 B
// 69632   : s0   32 x  32 bf16  (padded layer-0 input)       2048 B
// 71680   : p0   1024 x 32 bf16 (later orbU/orbD f32)       65536 B
// 137216  : p1   1024 x 32 bf16                             65536 B
// 202752  : red  64 f32 (su, lu, sd, ld)                      256 B
#define FN_SMEM 203008

__global__ __launch_bounds__(256, 1) void k_ferminet(FNParams P) {
    extern __shared__ __align__(16) char smem[];
    __bf16* sA   = (__bf16*)(smem + 0);
    __bf16* sTmp = (__bf16*)(smem + 53248);
    __bf16* wbuf = sTmp;                       // phase-2 overlay (32 x 256)
    __bf16* s0   = (__bf16*)(smem + 69632);
    __bf16* p0   = (__bf16*)(smem + 71680);
    __bf16* p1   = (__bf16*)(smem + 137216);
    float*  orbU = (float*)(smem + 71680);     // 16 x 256 f32 overlay
    float*  orbD = (float*)(smem + 88064);     // 16 x 256 f32 overlay
    float*  red  = (float*)(smem + 202752);    // su[16] lu[16] sd[16] ld[16]

    const int tid = threadIdx.x;
    const float* rb = P.r + (size_t)blockIdx.x * 96;   // 32 electrons x 3
    const float a0 = P.a[0], a1 = P.a[1], a2 = P.a[2];

    // ---- stage padded layer-0 inputs --------------------------------------
    for (int i = tid; i < 32 * 32; i += 256) s0[i] = (__bf16)0.0f;
    for (int i = tid; i < 1024 * 32; i += 256) p0[i] = (__bf16)0.0f;
    __syncthreads();
    if (tid < 32) {
        float dx = rb[tid * 3 + 0] - a0;
        float dy = rb[tid * 3 + 1] - a1;
        float dz = rb[tid * 3 + 2] - a2;
        float nr = sqrtf(dx * dx + dy * dy + dz * dz);
        s0[tid * 32 + 0] = (__bf16)dx;
        s0[tid * 32 + 1] = (__bf16)dy;
        s0[tid * 32 + 2] = (__bf16)dz;
        s0[tid * 32 + 3] = (__bf16)nr;
    }
    for (int idx = tid; idx < 1024; idx += 256) {
        int i = idx >> 5, j = idx & 31;
        float dx = rb[i * 3 + 0] - rb[j * 3 + 0];
        float dy = rb[i * 3 + 1] - rb[j * 3 + 1];
        float dz = rb[i * 3 + 2] - rb[j * 3 + 2];
        float sq = dx * dx + dy * dy + dz * dz;
        float dist = (sq > 0.0f) ? sqrtf(sq) : 0.0f;
        float e = (i == j) ? 1.0f : 0.0f;    // eye broadcast over all 4 ch
        p0[idx * 32 + 0] = (__bf16)(dx + e);
        p0[idx * 32 + 1] = (__bf16)(dy + e);
        p0[idx * 32 + 2] = (__bf16)(dz + e);
        p0[idx * 32 + 3] = (__bf16)(dist + e);
    }
    __syncthreads();

    // ---- 4 feature blocks -------------------------------------------------
    __bf16* pIn  = p0;
    __bf16* pOut = p1;
    for (int l = 0; l < 4; ++l) {
        if (l == 0)
            gemm_bf16(s0, 32, P.Wst[0], 32, P.bs[0], 2, 16, 1, sTmp, 256, 0);
        else
            gemm_bf16(sA, 832, P.Wst[l], 832, P.bs[l], 2, 16, 26, sTmp, 256, 0);
        gemm_bf16(pIn, 32, P.Wpt[l], 32, P.bp[l], 64, 2, 1, pOut, 32, 0);
        __syncthreads();

        // epilogue: copy h_s, spin means, masked pair means -> concat buffer
        {
            int c = tid;  // 256 threads == 256 columns
            float su = 0.0f, sd = 0.0f;
#pragma unroll
            for (int i = 0; i < 16; ++i) su += (float)sTmp[i * 256 + c];
#pragma unroll
            for (int i = 16; i < 32; ++i) sd += (float)sTmp[i * 256 + c];
            __bf16 mu = (__bf16)(su * 0.0625f);
            __bf16 md = (__bf16)(sd * 0.0625f);
            for (int i = 0; i < 32; ++i) {
                sA[i * 832 + c]       = sTmp[i * 256 + c];
                sA[i * 832 + 256 + c] = mu;
                sA[i * 832 + 512 + c] = md;
            }
        }
        for (int idx = tid; idx < 1024; idx += 256) {
            int j = idx >> 5, c = idx & 31;
            float gu = 0.0f, gd = 0.0f;
#pragma unroll
            for (int i = 0; i < 16; ++i) gu += (float)pOut[(i * 32 + j) * 32 + c];
#pragma unroll
            for (int i = 16; i < 32; ++i) gd += (float)pOut[(i * 32 + j) * 32 + c];
            sA[j * 832 + 768 + c] = (j >= 16) ? (__bf16)(gu * 0.03125f) : (__bf16)0.0f;
            sA[j * 832 + 800 + c] = (j < 16)  ? (__bf16)(gd * 0.03125f) : (__bf16)0.0f;
        }
        __syncthreads();
        __bf16* t = pIn; pIn = pOut; pOut = t;
    }

    // ---- w = tanh(dense(h_s_concat)) --------------------------------------
    gemm_bf16(sA, 832, P.Wwt, 832, P.bw, 2, 16, 26, wbuf, 256, 0);
    __syncthreads();

    // ---- orbitals: r_wu/r_wd (raw f32) ------------------------------------
    gemm_bf16(wbuf, 256, P.Wwut, 256, P.bwu, 1, 16, 8, orbU, 256, 1);
    gemm_bf16(wbuf + 16 * 256, 256, P.Wwdt, 256, P.bwd, 1, 16, 8, orbD, 256, 1);
    __syncthreads();

    // ---- multiply by exponential envelopes --------------------------------
    for (int idx = tid; idx < 4096; idx += 256) {
        int i = idx >> 8, c = idx & 255;
        float ux = rb[i * 3 + 0] - a0;
        float uy = rb[i * 3 + 1] - a1;
        float uz = rb[i * 3 + 2] - a2;
        float eu = expf(-(ux * P.euw[c] + uy * P.euw[256 + c] +
                          uz * P.euw[512 + c] + P.eub[c]));
        orbU[i * 256 + c] *= eu;
        float dxx = rb[(16 + i) * 3 + 0] - a0;
        float dyy = rb[(16 + i) * 3 + 1] - a1;
        float dzz = rb[(16 + i) * 3 + 2] - a2;
        float ed = expf(-(dxx * P.edw[c] + dyy * P.edw[256 + c] +
                          dzz * P.edw[512 + c] + P.edb[c]));
        orbD[i * 256 + c] *= ed;
    }
    __syncthreads();

    // ---- 32 slogdets: 16 lanes per determinant ----------------------------
    {
        int g = tid >> 4;     // 0..15, one 16x16 det per group per round
        int l = tid & 15;
        float s, ld;
        slogdet16(orbU + g * 256, l, s, ld);
        if (l == 0) { red[g] = s; red[16 + g] = ld; }
        slogdet16(orbD + g * 256, l, s, ld);
        if (l == 0) { red[32 + g] = s; red[48 + g] = ld; }
    }
    __syncthreads();

    if (tid == 0) {
        float mx = -3.4e38f;
#pragma unroll
        for (int k = 0; k < 16; ++k) {
            float t = red[16 + k] + red[48 + k];
            if (t > mx) mx = t;
        }
        float psi = 0.0f;
#pragma unroll
        for (int k = 0; k < 16; ++k)
            psi += red[k] * red[32 + k] * expf(red[16 + k] + red[48 + k] - mx);
        P.out[blockIdx.x] = logf(fabsf(psi)) + mx;
    }
}

// ---------------------------------------------------------------------------
extern "C" void kernel_launch(void* const* d_in, const int* in_sizes, int n_in,
                              void* d_out, int out_size, void* d_ws, size_t ws_size,
                              hipStream_t stream) {
    // input order (setup_inputs dict order): r, s[0..3]{w,b}, p[0..3]{w,b},
    // w{w,b}, wu{w,b}, wd{w,b}, env_u{w,b}, env_d{w,b}, a
    const float* r = (const float*)d_in[0];
    const float *sw[4], *sb[4], *pw[4], *pb[4];
    for (int l = 0; l < 4; ++l) {
        sw[l] = (const float*)d_in[1 + 2 * l];
        sb[l] = (const float*)d_in[2 + 2 * l];
        pw[l] = (const float*)d_in[9 + 2 * l];
        pb[l] = (const float*)d_in[10 + 2 * l];
    }
    const float* ww  = (const float*)d_in[17];
    const float* wb  = (const float*)d_in[18];
    const float* wuw = (const float*)d_in[19];
    const float* wub = (const float*)d_in[20];
    const float* wdw = (const float*)d_in[21];
    const float* wdb = (const float*)d_in[22];
    const float* euw = (const float*)d_in[23];
    const float* eub = (const float*)d_in[24];
    const float* edw = (const float*)d_in[25];
    const float* edb = (const float*)d_in[26];
    const float* a   = (const float*)d_in[27];

    __bf16* ws = (__bf16*)d_ws;
    size_t off = 0;
    auto prep = [&](const float* W, int K, int N, int Kpad) -> const __bf16* {
        __bf16* dst = ws + off;
        off += (size_t)N * Kpad;
        int total  = N * Kpad;
        int blocks = (total + 255) / 256;
        if (blocks > 1024) blocks = 1024;
        k_prep_bf16t<<<blocks, 256, 0, stream>>>(W, K, N, Kpad, dst);
        return dst;
    };

    FNParams P;
    P.r = r; P.a = a;
    P.Wst[0] = prep(sw[0], 4, 256, 32);
    for (int l = 1; l < 4; ++l) P.Wst[l] = prep(sw[l], 832, 256, 832);
    for (int l = 0; l < 4; ++l) P.bs[l] = sb[l];
    P.Wpt[0] = prep(pw[0], 4, 32, 32);
    for (int l = 1; l < 4; ++l) P.Wpt[l] = prep(pw[l], 32, 32, 32);
    for (int l = 0; l < 4; ++l) P.bp[l] = pb[l];
    P.Wwt  = prep(ww, 832, 256, 832);   P.bw  = wb;
    P.Wwut = prep(wuw, 256, 256, 256);  P.bwu = wub;
    P.Wwdt = prep(wdw, 256, 256, 256);  P.bwd = wdb;
    P.euw = euw; P.eub = eub; P.edw = edw; P.edb = edb;
    P.out = (float*)d_out;

    int nb = in_sizes[0] / 96;   // BATCH (= 2048)
    k_ferminet<<<nb, 256, FN_SMEM, stream>>>(P);
}
